// NA_ALISTA_54571854463057
// MI455X (gfx1250) — matrix operand order repaired and checked by reading, not compiled, not run on hardware
//
#include <hip/hip_runtime.h>
#include <hip/hip_bf16.h>
#include <math.h>

// ---------------- problem constants ----------------
#define BSZ   2048
#define MDIM  512
#define NDIM  2048
#define HDIM  128      // LSTM DIM
#define KITER 16

typedef __bf16 bhalf;
typedef __attribute__((ext_vector_type(16))) __bf16 bf16x16;
typedef __attribute__((ext_vector_type(8)))  float  f32x8;

union Frag  { bf16x16 v; float4 q[2]; };
union Half8 { float4 f; bhalf h[8]; };

// ============================================================
// Split-bf16 WMMA GEMM:  out = L(hi+lo) @ R(hi+lo)  (drop lo*lo)
// L: rows x K (row-major, hi/lo bf16), R: K x cols (row-major, hi/lo bf16)
// MODE 0: b = y - acc, store bf16 hi/lo of b
// MODE 1: store acc as fp32
// 256 threads = 8 waves arranged 4(M) x 2(N); BK = 32.
// Tile BM x BN templated: 128x128 -> 2x4 frags/wave, 64x64 -> 1x2.
// Register-double-buffered global->LDS staging; R transposed on the
// LDS-store side so global reads stay coalesced (16B contiguous chunks).
// ============================================================
template<int MODE, int BM, int BN>
__global__ __launch_bounds__(256) void na_gemm_wmma(
    const bhalf* __restrict__ Lhi, const bhalf* __restrict__ Llo,
    const bhalf* __restrict__ Rhi, const bhalf* __restrict__ Rlo,
    const float* __restrict__ Ymat,
    float* __restrict__ Cout,
    bhalf* __restrict__ Ohi, bhalf* __restrict__ Olo,
    int K, int cols)
{
    constexpr int BK  = 32;
    constexpr int FM  = BM / 64;          // fragments per wave in M (4 waves in M)
    constexpr int FN  = BN / 32;          // fragments per wave in N (2 waves in N)
    constexpr int LF4 = BM / 64;          // 8-half chunks per thread for L tile
    constexpr int RF4 = BN / 64;          // 8-half chunks per thread for R tile
    constexpr int RSEGROW = BN / 8;       // 8-half segments per R row

    __shared__ __align__(16) bhalf sLhi[BM * BK];
    __shared__ __align__(16) bhalf sLlo[BM * BK];
    __shared__ __align__(16) bhalf sRhi[BN * BK];   // transposed: [n][k]
    __shared__ __align__(16) bhalf sRlo[BN * BK];

    const int tid  = threadIdx.x;
    const int lane = tid & 31;
    const int wave = tid >> 5;
    const int wm   = wave >> 1;          // 0..3
    const int wn   = wave & 1;           // 0..1
    const int rowBase = blockIdx.y * BM;
    const int colBase = blockIdx.x * BN;

    f32x8 acc[FM][FN] = {};

    // prefetch registers for next tile
    Half8 lph[LF4], lpl[LF4];
    Half8 rph[RF4], rpl[RF4];

    auto load_global = [&](int kb) {
        #pragma unroll
        for (int i = 0; i < LF4; ++i) {
            int g  = tid * LF4 + i;       // 8-half chunk id
            int r  = g >> 2;              // 4 chunks per 32-half row
            int ck = (g & 3) * 8;
            size_t gi = (size_t)(rowBase + r) * K + kb + ck;
            lph[i].f = *reinterpret_cast<const float4*>(Lhi + gi);
            lpl[i].f = *reinterpret_cast<const float4*>(Llo + gi);
        }
        #pragma unroll
        for (int i = 0; i < RF4; ++i) {
            int s  = tid * RF4 + i;
            int k  = s / RSEGROW;
            int c0 = (s % RSEGROW) * 8;
            size_t gi = (size_t)(kb + k) * cols + colBase + c0;
            rph[i].f = *reinterpret_cast<const float4*>(Rhi + gi);
            rpl[i].f = *reinterpret_cast<const float4*>(Rlo + gi);
        }
    };

    auto store_lds = [&]() {
        #pragma unroll
        for (int i = 0; i < LF4; ++i) {
            int g  = tid * LF4 + i;
            int r  = g >> 2;
            int ck = (g & 3) * 8;
            *reinterpret_cast<float4*>(sLhi + r * BK + ck) = lph[i].f;
            *reinterpret_cast<float4*>(sLlo + r * BK + ck) = lpl[i].f;
        }
        #pragma unroll
        for (int i = 0; i < RF4; ++i) {
            int s  = tid * RF4 + i;
            int k  = s / RSEGROW;
            int c0 = (s % RSEGROW) * 8;
            #pragma unroll
            for (int j = 0; j < 8; ++j) {   // transpose on store
                sRhi[(c0 + j) * BK + k] = rph[i].h[j];
                sRlo[(c0 + j) * BK + k] = rpl[i].h[j];
            }
        }
    };

    load_global(0);

    for (int kb = 0; kb < K; kb += BK) {
        store_lds();
        __syncthreads();
        if (kb + BK < K) load_global(kb + BK);   // overlap with WMMA compute

        // ---- B fragments: lane n = lane&15, K half = lane>>4, 16 contiguous k ----
        Frag bhiF[FN], bloF[FN];
        #pragma unroll
        for (int fn = 0; fn < FN; ++fn) {
            int n  = wn * (FN * 16) + fn * 16 + (lane & 15);
            int kh = (lane >> 4) * 16;
            const float4* ph = reinterpret_cast<const float4*>(sRhi + n * BK + kh);
            const float4* pl = reinterpret_cast<const float4*>(sRlo + n * BK + kh);
            bhiF[fn].q[0] = ph[0]; bhiF[fn].q[1] = ph[1];
            bloF[fn].q[0] = pl[0]; bloF[fn].q[1] = pl[1];
        }
        // ---- A fragments: m = lane&15, bk = (lane>>4)*16,
        //      16 contiguous k [bk..bk+15] ----
        #pragma unroll
        for (int fm = 0; fm < FM; ++fm) {
            int m  = wm * (FM * 16) + fm * 16 + (lane & 15);
            int bk = (lane >> 4) * 16;
            Frag ahiF, aloF;
            ahiF.q[0] = *reinterpret_cast<const float4*>(sLhi + m * BK + bk);
            ahiF.q[1] = *reinterpret_cast<const float4*>(sLhi + m * BK + bk + 8);
            aloF.q[0] = *reinterpret_cast<const float4*>(sLlo + m * BK + bk);
            aloF.q[1] = *reinterpret_cast<const float4*>(sLlo + m * BK + bk + 8);
            #pragma unroll
            for (int fn = 0; fn < FN; ++fn) {
                acc[fm][fn] = __builtin_amdgcn_wmma_f32_16x16x32_bf16(
                    false, ahiF.v, false, bhiF[fn].v, (short)0, acc[fm][fn], false, false);
                acc[fm][fn] = __builtin_amdgcn_wmma_f32_16x16x32_bf16(
                    false, ahiF.v, false, bloF[fn].v, (short)0, acc[fm][fn], false, false);
                acc[fm][fn] = __builtin_amdgcn_wmma_f32_16x16x32_bf16(
                    false, aloF.v, false, bhiF[fn].v, (short)0, acc[fm][fn], false, false);
            }
        }
        __syncthreads();
    }

    // ---- epilogue: C/D layout M = r + 8*(lane>>4), N = lane&15 ----
    const int mo = (lane >> 4) * 8;
    const int nc = lane & 15;
    #pragma unroll
    for (int fm = 0; fm < FM; ++fm)
        #pragma unroll
        for (int fn = 0; fn < FN; ++fn)
            #pragma unroll
            for (int r = 0; r < 8; ++r) {
                int row = rowBase + wm * (FM * 16) + fm * 16 + r + mo;
                int col = colBase + wn * (FN * 16) + fn * 16 + nc;
                size_t idx = (size_t)row * cols + col;
                float v = acc[fm][fn][r];
                if constexpr (MODE == 0) {
                    float bv = Ymat[idx] - v;
                    bhalf h = (bhalf)bv;
                    Ohi[idx] = h;
                    Olo[idx] = (bhalf)(bv - (float)h);
                } else {
                    Cout[idx] = v;
                }
            }
}

// ============================================================
// per-row L1 reductions
// ============================================================
__global__ __launch_bounds__(256) void na_rowl1_bf16(
    const bhalf* __restrict__ hi, const bhalf* __restrict__ lo,
    float* __restrict__ out, int len)
{
    int row = blockIdx.x, t = threadIdx.x;
    float s = 0.f;
    for (int j = t; j < len; j += 256)
        s += fabsf((float)hi[(size_t)row * len + j] + (float)lo[(size_t)row * len + j]);
    __shared__ float red[256];
    red[t] = s; __syncthreads();
    for (int k = 128; k > 0; k >>= 1) { if (t < k) red[t] += red[t + k]; __syncthreads(); }
    if (t == 0) out[row] = red[0];
}

__global__ __launch_bounds__(256) void na_rowl1_f32(
    const float* __restrict__ src, float* __restrict__ out, int len)
{
    int row = blockIdx.x, t = threadIdx.x;
    float s = 0.f;
    for (int j = t; j < len; j += 256) s += fabsf(src[(size_t)row * len + j]);
    __shared__ float red[256];
    red[t] = s; __syncthreads();
    for (int k = 128; k > 0; k >>= 1) { if (t < k) red[t] += red[t + k]; __syncthreads(); }
    if (t == 0) out[row] = red[0];
}

// ============================================================
// batch stats (iteration 0): mean & sample std (ddof=1) of l1_b, l1_c
// ============================================================
__global__ __launch_bounds__(256) void na_stats(
    const float* __restrict__ l1b, const float* __restrict__ l1c,
    float* __restrict__ stats, int n)
{
    int t = threadIdx.x;
    float sb = 0.f, qb = 0.f, sc = 0.f, qc = 0.f;
    for (int j = t; j < n; j += 256) {
        float a = l1b[j]; sb += a; qb += a * a;
        float b = l1c[j]; sc += b; qc += b * b;
    }
    __shared__ float r0[256], r1[256], r2[256], r3[256];
    r0[t] = sb; r1[t] = qb; r2[t] = sc; r3[t] = qc; __syncthreads();
    for (int k = 128; k > 0; k >>= 1) {
        if (t < k) { r0[t]+=r0[t+k]; r1[t]+=r1[t+k]; r2[t]+=r2[t+k]; r3[t]+=r3[t+k]; }
        __syncthreads();
    }
    if (t == 0) {
        float fn = (float)n;
        float mb = r0[0] / fn;
        float vb = (r1[0] - fn * mb * mb) / (fn - 1.f);
        float mc = r2[0] / fn;
        float vc = (r3[0] - fn * mc * mc) / (fn - 1.f);
        stats[0] = mb; stats[1] = sqrtf(fmaxf(vb, 0.f));
        stats[2] = mc; stats[3] = sqrtf(fmaxf(vc, 0.f));
    }
}

// ============================================================
// LSTM cell + MLP head, one block (128 threads) per batch row
// ============================================================
__device__ __forceinline__ float na_sigmoid(float x) { return 1.f / (1.f + expf(-x)); }
__device__ __forceinline__ float na_softplus(float x) { return (x > 30.f) ? x : log1pf(expf(x)); }

__global__ __launch_bounds__(128) void na_lstm(
    const float* __restrict__ l1b, const float* __restrict__ l1c,
    const float* __restrict__ stats,
    const float* __restrict__ W_ih, const float* __restrict__ b_ih,
    const float* __restrict__ W_hh, const float* __restrict__ b_hh,
    const float* __restrict__ W_lll, const float* __restrict__ b_lll,
    const float* __restrict__ W_lin, const float* __restrict__ b_lin,
    float* __restrict__ hidden, float* __restrict__ cell,
    float* __restrict__ gamma, float* __restrict__ theta)
{
    int row = blockIdx.x, t = threadIdx.x;
    __shared__ float sh[HDIM], sc2[HDIM], sr[HDIM], red0[HDIM], red1[HDIM];

    float st0 = (l1b[row] - stats[0]) / stats[1];
    float st1 = (l1c[row] - stats[2]) / stats[3];
    sh[t] = hidden[(size_t)row * HDIM + t];
    float cprev = cell[(size_t)row * HDIM + t];
    __syncthreads();

    float g4[4];
    #pragma unroll
    for (int q = 0; q < 4; ++q) {
        int gi = q * HDIM + t;
        float acc = b_ih[gi] + b_hh[gi] + W_ih[gi * 2] * st0 + W_ih[gi * 2 + 1] * st1;
        const float* wr = &W_hh[(size_t)gi * HDIM];
        for (int j = 0; j < HDIM; ++j) acc = fmaf(wr[j], sh[j], acc);
        g4[q] = acc;
    }
    float i_ = na_sigmoid(g4[0]);
    float f_ = na_sigmoid(g4[1]);
    float g_ = tanhf(g4[2]);
    float o_ = na_sigmoid(g4[3]);
    float c2 = f_ * cprev + i_ * g_;
    float h2 = o_ * tanhf(c2);
    cell[(size_t)row * HDIM + t] = c2;
    hidden[(size_t)row * HDIM + t] = h2;
    sc2[t] = c2;
    __syncthreads();

    float r = b_lll[t];
    const float* wl = &W_lll[(size_t)t * HDIM];
    for (int j = 0; j < HDIM; ++j) r = fmaf(wl[j], sc2[j], r);
    sr[t] = fmaxf(r, 0.f);
    __syncthreads();

    red0[t] = W_lin[t] * sr[t];
    red1[t] = W_lin[HDIM + t] * sr[t];
    __syncthreads();
    for (int k = 64; k > 0; k >>= 1) {
        if (t < k) { red0[t] += red0[t + k]; red1[t] += red1[t + k]; }
        __syncthreads();
    }
    if (t == 0) {
        gamma[row] = na_softplus(red0[0] + b_lin[0]);
        theta[row] = na_softplus(red1[0] + b_lin[1]);
    }
}

// ============================================================
// d = x + gamma*c ; radix-select p-th largest |d| per row ;
// x = (|d| > thresh) ? d : sign(d)*relu(|d|-theta) ; write fp32 + bf16 hi/lo
// ============================================================
__global__ __launch_bounds__(256) void na_thresh(
    const float* __restrict__ c, const float* __restrict__ gamma,
    const float* __restrict__ theta,
    float* __restrict__ x, bhalf* __restrict__ xhi, bhalf* __restrict__ xlo,
    int p, int len)
{
    int row = blockIdx.x, t = threadIdx.x;
    float gm = gamma[row], th = theta[row];

    float d[8]; unsigned key[8];
    #pragma unroll
    for (int e = 0; e < 8; ++e) {
        int j = t + e * 256;
        size_t idx = (size_t)row * len + j;
        float dv = x[idx] + gm * c[idx];
        d[e] = dv;
        key[e] = __float_as_uint(fabsf(dv));   // non-negative float: bit order == value order
    }

    __shared__ unsigned hist[256];
    __shared__ unsigned s_pref;
    __shared__ int s_rem;
    if (t == 0) { s_pref = 0u; s_rem = p; }
    __syncthreads();

    for (int round = 0; round < 4; ++round) {
        int shift = 24 - 8 * round;
        unsigned knownMask = (round == 0) ? 0u : (0xFFFFFFFFu << (shift + 8));
        unsigned pref = s_pref;
        hist[t] = 0u;
        __syncthreads();
        #pragma unroll
        for (int e = 0; e < 8; ++e)
            if ((key[e] & knownMask) == pref)
                atomicAdd(&hist[(key[e] >> shift) & 255u], 1u);
        __syncthreads();
        if (t == 0) {
            int rem = s_rem; unsigned pr = s_pref;
            for (int b = 255; b >= 0; --b) {
                int cnt = (int)hist[b];
                if (rem <= cnt) { pr |= ((unsigned)b) << shift; break; }
                rem -= cnt;
            }
            s_pref = pr; s_rem = rem;
        }
        __syncthreads();
    }
    float thresh = __uint_as_float(s_pref);

    #pragma unroll
    for (int e = 0; e < 8; ++e) {
        int j = t + e * 256;
        size_t idx = (size_t)row * len + j;
        float dv = d[e], ad = fabsf(dv);
        float xn = (ad > thresh) ? dv : copysignf(fmaxf(ad - th, 0.f), dv);
        x[idx] = xn;
        bhalf h = (bhalf)xn;
        xhi[idx] = h;
        xlo[idx] = (bhalf)(xn - (float)h);
    }
}

// ============================================================
// setup kernels
// ============================================================
__global__ __launch_bounds__(256) void na_prep_A(
    const float* __restrict__ A,
    bhalf* __restrict__ Ahi, bhalf* __restrict__ Alo,
    bhalf* __restrict__ Athi, bhalf* __restrict__ Atlo)
{
    int idx = blockIdx.x * 256 + threadIdx.x;
    if (idx >= MDIM * NDIM) return;
    int m = idx >> 11, n = idx & (NDIM - 1);
    float a = A[idx];
    bhalf h = (bhalf)a;
    bhalf l = (bhalf)(a - (float)h);
    Ahi[idx] = h; Alo[idx] = l;
    Athi[(size_t)n * MDIM + m] = h;
    Atlo[(size_t)n * MDIM + m] = l;
}

__global__ __launch_bounds__(256) void na_init_x(
    float* __restrict__ x, bhalf* __restrict__ xhi, bhalf* __restrict__ xlo, size_t n)
{
    size_t i = (size_t)blockIdx.x * 256 + threadIdx.x;
    if (i < n) { x[i] = 0.f; xhi[i] = (bhalf)0.f; xlo[i] = (bhalf)0.f; }
}

__global__ __launch_bounds__(256) void na_init_state(
    const float* __restrict__ h0, const float* __restrict__ c0,
    float* __restrict__ hidden, float* __restrict__ cell, int total)
{
    int i = blockIdx.x * 256 + threadIdx.x;
    if (i < total) { int j = i & (HDIM - 1); hidden[i] = h0[j]; cell[i] = c0[j]; }
}

// ============================================================
// host launcher
// ============================================================
extern "C" void kernel_launch(void* const* d_in, const int* in_sizes, int n_in,
                              void* d_out, int out_size, void* d_ws, size_t ws_size,
                              hipStream_t stream)
{
    const float* y     = (const float*)d_in[0];
    const float* A     = (const float*)d_in[1];
    const float* W_ih  = (const float*)d_in[2];
    const float* W_hh  = (const float*)d_in[3];
    const float* b_ih  = (const float*)d_in[4];
    const float* b_hh  = (const float*)d_in[5];
    const float* W_lll = (const float*)d_in[6];
    const float* b_lll = (const float*)d_in[7];
    const float* W_lin = (const float*)d_in[8];
    const float* b_lin = (const float*)d_in[9];
    const float* h0    = (const float*)d_in[10];
    const float* c0    = (const float*)d_in[11];

    // ---- workspace carve-up (256B aligned) ----
    char* base = (char*)d_ws;
    size_t off = 0;
    auto carve = [&](size_t bytes) -> char* {
        char* p = base + off;
        off = (off + bytes + 255) & ~(size_t)255;
        return p;
    };
    bhalf* Ahi  = (bhalf*)carve((size_t)MDIM * NDIM * 2);
    bhalf* Alo  = (bhalf*)carve((size_t)MDIM * NDIM * 2);
    bhalf* Athi = (bhalf*)carve((size_t)NDIM * MDIM * 2);
    bhalf* Atlo = (bhalf*)carve((size_t)NDIM * MDIM * 2);
    float* xbuf = (float*)carve((size_t)BSZ * NDIM * 4);
    bhalf* xhi  = (bhalf*)carve((size_t)BSZ * NDIM * 2);
    bhalf* xlo  = (bhalf*)carve((size_t)BSZ * NDIM * 2);
    bhalf* bhi  = (bhalf*)carve((size_t)BSZ * MDIM * 2);
    bhalf* blo  = (bhalf*)carve((size_t)BSZ * MDIM * 2);
    float* cbuf = (float*)carve((size_t)BSZ * NDIM * 4);
    float* l1b  = (float*)carve((size_t)BSZ * 4);
    float* l1c  = (float*)carve((size_t)BSZ * 4);
    float* stats  = (float*)carve(16);
    float* hidden = (float*)carve((size_t)BSZ * HDIM * 4);
    float* cell   = (float*)carve((size_t)BSZ * HDIM * 4);
    float* gammaA = (float*)carve((size_t)BSZ * 4);
    float* thetaA = (float*)carve((size_t)BSZ * 4);
    (void)ws_size; (void)n_in; (void)in_sizes;

    // s_sched = np.clip(np.linspace(S/K, 1.4*S, K), 0, 1.2*S).astype(int32)
    int sched[KITER];
    for (int i = 0; i < KITER; ++i) {
        double v = (double)i * (133.75 / 15.0) + 6.25;
        if (v > 120.0) v = 120.0;
        if (v < 0.0) v = 0.0;
        sched[i] = (int)v;
    }

    // ---- setup ----
    na_prep_A<<<(MDIM * NDIM + 255) / 256, 256, 0, stream>>>(A, Ahi, Alo, Athi, Atlo);
    {
        size_t n = (size_t)BSZ * NDIM;
        na_init_x<<<(unsigned)((n + 255) / 256), 256, 0, stream>>>(xbuf, xhi, xlo, n);
    }
    na_init_state<<<(BSZ * HDIM + 255) / 256, 256, 0, stream>>>(h0, c0, hidden, cell, BSZ * HDIM);

    // ---- main K-iteration loop ----
    for (int it = 0; it < KITER; ++it) {
        // b = y - x @ A^T   (rows=B, cols=M, K=N) : 64x64 tiles -> 8x32 = 256 WGs
        na_gemm_wmma<0, 64, 64><<<dim3(MDIM / 64, BSZ / 64), 256, 0, stream>>>(
            xhi, xlo, Athi, Atlo, y, nullptr, bhi, blo, NDIM, MDIM);
        // c = b @ A         (rows=B, cols=N, K=M) : 128x128 tiles -> 16x16 = 256 WGs
        na_gemm_wmma<1, 128, 128><<<dim3(NDIM / 128, BSZ / 128), 256, 0, stream>>>(
            bhi, blo, Ahi, Alo, nullptr, cbuf, nullptr, nullptr, MDIM, NDIM);

        na_rowl1_bf16<<<BSZ, 256, 0, stream>>>(bhi, blo, l1b, MDIM);
        na_rowl1_f32 <<<BSZ, 256, 0, stream>>>(cbuf, l1c, NDIM);

        if (it == 0)
            na_stats<<<1, 256, 0, stream>>>(l1b, l1c, stats, BSZ);

        na_lstm<<<BSZ, HDIM, 0, stream>>>(l1b, l1c, stats,
                                          W_ih, b_ih, W_hh, b_hh,
                                          W_lll, b_lll, W_lin, b_lin,
                                          hidden, cell, gammaA, thetaA);

        na_thresh<<<BSZ, 256, 0, stream>>>(cbuf, gammaA, thetaA,
                                           xbuf, xhi, xlo, sched[it], NDIM);
    }

    hipMemcpyAsync(d_out, xbuf, (size_t)BSZ * NDIM * sizeof(float),
                   hipMemcpyDeviceToDevice, stream);
    (void)out_size;
}
